// BiLSTM_19404662243425
// MI455X (gfx1250) — matrix operand order repaired and checked by reading, not compiled, not run on hardware
//
#include <hip/hip_runtime.h>

// ---------------------------------------------------------------------------
// LSTM language-model decode for gfx1250 (MI455X), bf16 WMMA matrix cores.
// 2x2 register-blocked WMMA GEMM, ping-pong (unroll-2) pipelined fragment
// loads with fixed register sets (no rotating copies -> no spills), and
// WGP-scope global_prefetch_b8 via inline asm for the weight stream.
// ---------------------------------------------------------------------------

typedef __attribute__((ext_vector_type(16))) __bf16 v16bf;
typedef __attribute__((ext_vector_type(8)))  float  v8f;
typedef __attribute__((ext_vector_type(4)))  float  v4f;

union Frag16 { v16bf bf; v4f f4[2]; };

#define VOCAB 32000
#define EMB   300
#define HID   512
#define COLOR 54
#define SEQ   20
#define BATCH 128
#define NGATE 2048          // 4*HID
#define KGATE 896           // [0,354): x  [354,360): pad  [360,872): h  [872,896): pad
#define HOFF  360           // h offset (16B aligned: 360*2 = 720 bytes)

__device__ __forceinline__ float sigf(float x) { return 1.0f / (1.0f + __expf(-x)); }

// A fragment (16x32, contiguous k order chosen here, not the ISA table order): per lane 16 contiguous K values
__device__ __forceinline__ void loadA(Frag16& f, const __bf16* p, int k) {
  f.f4[0] = *(const v4f*)(p + k);
  f.f4[1] = *(const v4f*)(p + k + 8);
}
// B fragment (32x16, same contiguous k order as A, not the ISA table order): per lane 16 contiguous K values
__device__ __forceinline__ void loadB(Frag16& f, const __bf16* p, int k) {
  f.f4[0] = *(const v4f*)(p + k);
  f.f4[1] = *(const v4f*)(p + k + 8);
}
__device__ __forceinline__ v8f wmma_bf16(const Frag16& a, const Frag16& b, v8f c) {
  return __builtin_amdgcn_wmma_f32_16x16x32_bf16(false, a.bf, false, b.bf,
                                                 (short)0, c, false, false);
}
// WGP-scope prefetch (default scope 0 = WGP: pulls into all cache levels).
__device__ __forceinline__ void prefetch_wgp(const void* p) {
  asm volatile("global_prefetch_b8 %0, off offset:256" :: "v"(p));
}

// ---- one-time weight conversion -------------------------------------------
__global__ void cvt_bf16_kernel(const float* __restrict__ in, __bf16* __restrict__ out, int n) {
  int i = blockIdx.x * blockDim.x + threadIdx.x;
  if (i < n) out[i] = (__bf16)in[i];
}

__global__ void build_wcomb_kernel(const float* __restrict__ w_ih,
                                   const float* __restrict__ w_hh,
                                   __bf16* __restrict__ wc) {
  int i = blockIdx.x * blockDim.x + threadIdx.x;
  if (i >= NGATE * KGATE) return;
  int n = i / KGATE, k = i % KGATE;
  float v = 0.0f;
  if (k < COLOR + EMB)                  v = w_ih[n * (COLOR + EMB) + k];
  else if (k >= HOFF && k < HOFF + HID) v = w_hh[n * HID + (k - HOFF)];
  wc[i] = (__bf16)v;
}

__global__ void build_bias_kernel(const float* __restrict__ a, const float* __restrict__ b,
                                  float* __restrict__ o, int n) {
  int i = blockIdx.x * blockDim.x + threadIdx.x;
  if (i < n) o[i] = a[i] + b[i];
}

// ---- initial state: h=c=0, x0 = [color | emb(tok0)] -----------------------
__global__ void init_state_kernel(const int* __restrict__ seq, const float* __restrict__ color,
                                  const float* __restrict__ emb, __bf16* __restrict__ xh,
                                  float* __restrict__ c, float* __restrict__ nll) {
  int b = blockIdx.x, tid = threadIdx.x;
  if (b == 0 && tid == 0) *nll = 0.0f;
  for (int j = tid; j < HID; j += blockDim.x) c[b * HID + j] = 0.0f;
  for (int i = tid; i < COLOR; i += blockDim.x)
    xh[(size_t)b * KGATE + i] = (__bf16)color[b * COLOR + i];
  int tok = seq[b];  // sequence_batch[0][b]
  for (int e = tid; e < EMB; e += blockDim.x)
    xh[(size_t)b * KGATE + COLOR + e] = (__bf16)emb[(size_t)tok * EMB + e];
  for (int j = COLOR + EMB + tid; j < KGATE; j += blockDim.x)   // zero h + pads
    xh[(size_t)b * KGATE + j] = (__bf16)0.0f;
}

// ---- bf16 WMMA GEMM:  C[M,N] = A[M,K] * W[N,K]^T + bias -------------------
// 2x2 register blocking (wave computes 32Mx32N, 4 accumulators).
// K loop unrolled by 2 with two fixed fragment sets (ping-pong): set1 loads
// for k+32 are issued before the set0 WMMAs, set0 loads for k+64 before the
// set1 WMMAs. The trailing set0 load at k==K is dead (reads <=64B past the
// row end inside the workspace; values never consumed).
// Requires M%32==0, N%32==0, K%64==0. All loads 16B-aligned.
__global__ void __launch_bounds__(256, 1)
wmma_gemm_bf16_kernel(const __bf16* __restrict__ A, int lda,
                      const __bf16* __restrict__ W, int ldw,
                      const float* __restrict__ bias,
                      float* __restrict__ C, int M, int N, int K) {
  const int lane = threadIdx.x & 31;
  const int wave = threadIdx.x >> 5;
  const int wpb  = blockDim.x >> 5;
  const int Mt2 = M >> 5, Nt2 = N >> 5;
  const int tiles = Mt2 * Nt2;
  const int l15 = lane & 15;
  const int alo = (lane < 16) ? 0 : 16;   // A half-lane K split
  const int blo = (lane < 16) ? 0 : 16;   // B half-lane K split

  for (int t = blockIdx.x * wpb + wave; t < tiles; t += gridDim.x * wpb) {
    int m0 = (t % Mt2) << 5;              // mt fast => waves in block share B rows
    int n0 = (t / Mt2) << 5;
    const __bf16* A0 = A + (size_t)(m0 + l15) * lda + alo;
    const __bf16* A1 = A0 + (size_t)16 * lda;
    const __bf16* B0 = W + (size_t)(n0 + l15) * ldw + blo;
    const __bf16* B1 = B0 + (size_t)16 * ldw;

    Frag16 a0, a1, b0, b1;    // set0
    Frag16 a2, a3, b2, b3;    // set1
    loadA(a0, A0, 0); loadA(a1, A1, 0);
    loadB(b0, B0, 0); loadB(b1, B1, 0);
    v8f c00 = {}, c01 = {}, c10 = {}, c11 = {};

    for (int k = 0; k < K; k += 64) {
      prefetch_wgp(B0 + k);               // +256B ahead of the B stream (WGP scope)
      prefetch_wgp(B1 + k);
      loadA(a2, A0, k + 32); loadA(a3, A1, k + 32);
      loadB(b2, B0, k + 32); loadB(b3, B1, k + 32);
      c00 = wmma_bf16(a0, b0, c00);
      c01 = wmma_bf16(a0, b1, c01);
      c10 = wmma_bf16(a1, b0, c10);
      c11 = wmma_bf16(a1, b1, c11);
      loadA(a0, A0, k + 64); loadA(a1, A1, k + 64);   // dead on last iteration
      loadB(b0, B0, k + 64); loadB(b1, B1, k + 64);
      c00 = wmma_bf16(a2, b2, c00);
      c01 = wmma_bf16(a2, b3, c01);
      c10 = wmma_bf16(a3, b2, c10);
      c11 = wmma_bf16(a3, b3, c11);
    }

    const int mrB = m0 + ((lane < 16) ? 0 : 8);
    const int nB  = n0 + l15;
#pragma unroll
    for (int mt = 0; mt < 2; ++mt) {
#pragma unroll
      for (int nt = 0; nt < 2; ++nt) {
        const v8f& acc = (mt == 0) ? (nt == 0 ? c00 : c01) : (nt == 0 ? c10 : c11);
        int   n  = nB + nt * 16;
        float bv = bias ? bias[n] : 0.0f;
        float* Cp = C + (size_t)(mrB + mt * 16) * N + n;
#pragma unroll
        for (int r = 0; r < 8; ++r) Cp[(size_t)r * N] = acc[r] + bv;
      }
    }
  }
}

// ---- LSTM cell update: c,h from gates; write bf16 h back into xh ----------
__global__ void lstm_cell_kernel(const float* __restrict__ gates,
                                 float* __restrict__ c, __bf16* __restrict__ xh) {
  int i = blockIdx.x * blockDim.x + threadIdx.x;
  if (i >= BATCH * HID) return;
  int b = i >> 9, j = i & (HID - 1);
  const float* g = gates + (size_t)b * NGATE;
  float ig = sigf(g[j]);
  float fg = sigf(g[HID + j]);
  float gg = tanhf(g[2 * HID + j]);
  float og = sigf(g[3 * HID + j]);
  float cn = fg * c[i] + ig * gg;
  c[i] = cn;
  xh[(size_t)b * KGATE + HOFF + j] = (__bf16)(og * tanhf(cn));
}

// ---- per-row argmax + logsumexp + NLL + next-token embedding --------------
#define RED_T 512
__global__ void argmax_nll_embed_kernel(const float* __restrict__ logits,
                                        const int* __restrict__ seq,
                                        const float* __restrict__ emb,
                                        __bf16* __restrict__ xh,
                                        float* __restrict__ nll_acc,
                                        float* __restrict__ pred_out, int t) {
  __shared__ float sval[RED_T];
  __shared__ int   sidx[RED_T];
  int b = blockIdx.x, tid = threadIdx.x;
  const float* row = logits + (size_t)b * VOCAB;

  float bm = -3.4e38f; int bi = 0;
  for (int v = tid; v < VOCAB; v += RED_T) {
    float x = row[v];
    if (x > bm) { bm = x; bi = v; }        // strided scan => first-max within thread
  }
  sval[tid] = bm; sidx[tid] = bi;
  __syncthreads();
  for (int s = RED_T / 2; s > 0; s >>= 1) {
    if (tid < s) {
      float ov = sval[tid + s]; int oi = sidx[tid + s];
      if (ov > sval[tid] || (ov == sval[tid] && oi < sidx[tid])) { sval[tid] = ov; sidx[tid] = oi; }
    }
    __syncthreads();
  }
  float maxv = sval[0]; int amax = sidx[0];
  __syncthreads();

  float sum = 0.0f;
  for (int v = tid; v < VOCAB; v += RED_T) sum += __expf(row[v] - maxv);
  sval[tid] = sum;
  __syncthreads();
  for (int s = RED_T / 2; s > 0; s >>= 1) {
    if (tid < s) sval[tid] += sval[tid + s];
    __syncthreads();
  }
  if (tid == 0) {
    int label = seq[(t + 1) * BATCH + b];            // sequence_batch[t+1][b]
    atomicAdd(nll_acc, maxv + logf(sval[0]) - row[label]);
    pred_out[t * BATCH + b] = (float)amax;
  }
  for (int e = tid; e < EMB; e += RED_T)             // feed prediction forward
    xh[(size_t)b * KGATE + COLOR + e] = (__bf16)emb[(size_t)amax * EMB + e];
}

__global__ void finalize_kernel(const float* __restrict__ nll, float* __restrict__ out) {
  out[0] = *nll / (float)((SEQ - 1) * BATCH);
}

// ---------------------------------------------------------------------------
extern "C" void kernel_launch(void* const* d_in, const int* in_sizes, int n_in,
                              void* d_out, int out_size, void* d_ws, size_t ws_size,
                              hipStream_t stream) {
  const int*   seq   = (const int*)  d_in[0];
  // d_in[1] = sequence_lengths (unused by the reference)
  const float* color = (const float*)d_in[2];
  const float* emb   = (const float*)d_in[3];
  const float* w_ih  = (const float*)d_in[4];
  const float* w_hh  = (const float*)d_in[5];
  const float* b_ih  = (const float*)d_in[6];
  const float* b_hh  = (const float*)d_in[7];
  const float* w_out = (const float*)d_in[8];
  const float* b_out = (const float*)d_in[9];
  float* out = (float*)d_out;   // [0] = loss, [1..] = predictions (19*128) as floats

  char* base = (char*)d_ws;
  size_t off = 0;
  auto take = [&](size_t bytes) -> char* {
    char* p = base + off;
    off += (bytes + 255) & ~(size_t)255;
    return p;
  };
  __bf16* wout_bf = (__bf16*)take((size_t)VOCAB * HID * 2);   // 32.8 MB (L2 resident)
  __bf16* wcomb   = (__bf16*)take((size_t)NGATE * KGATE * 2); // 3.7 MB
  float*  bcomb   = (float*) take((size_t)NGATE * 4);
  __bf16* xh      = (__bf16*)take((size_t)BATCH * KGATE * 2); // [x|h] bf16 input
  float*  cstate  = (float*) take((size_t)BATCH * HID * 4);
  float*  gates   = (float*) take((size_t)BATCH * NGATE * 4);
  float*  logits  = (float*) take((size_t)BATCH * VOCAB * 4); // 16.4 MB
  float*  nll     = (float*) take(256);
  (void)ws_size; (void)in_sizes; (void)n_in; (void)out_size;

  // one-time precompute (bf16 weight images, fused bias)
  cvt_bf16_kernel<<<(VOCAB * HID + 255) / 256, 256, 0, stream>>>(w_out, wout_bf, VOCAB * HID);
  build_wcomb_kernel<<<(NGATE * KGATE + 255) / 256, 256, 0, stream>>>(w_ih, w_hh, wcomb);
  build_bias_kernel<<<(NGATE + 255) / 256, 256, 0, stream>>>(b_ih, b_hh, bcomb, NGATE);
  init_state_kernel<<<BATCH, 256, 0, stream>>>(seq, color, emb, xh, cstate, nll);

  for (int t = 0; t < SEQ - 1; ++t) {
    // gates[128,2048] = xh[128,896] @ wcomb^T + (b_ih + b_hh)
    // tiles = (128/32)*(2048/32) = 256 wave-tiles -> 32 blocks * 8 waves
    wmma_gemm_bf16_kernel<<<32, 256, 0, stream>>>(xh, KGATE, wcomb, KGATE, bcomb,
                                                  gates, BATCH, NGATE, KGATE);
    lstm_cell_kernel<<<(BATCH * HID + 255) / 256, 256, 0, stream>>>(gates, cstate, xh);
    // logits[128,32000] = h[128,512] @ w_out^T + b_out   (dominant GEMM)
    // tiles = (128/32)*(32000/32) = 4000 wave-tiles -> 500 blocks * 8 waves
    wmma_gemm_bf16_kernel<<<500, 256, 0, stream>>>(xh + HOFF, KGATE, wout_bf, HID, b_out,
                                                   logits, BATCH, VOCAB, HID);
    argmax_nll_embed_kernel<<<BATCH, RED_T, 0, stream>>>(logits, seq, emb, xh, nll, out + 1, t);
  }
  finalize_kernel<<<1, 1, 0, stream>>>(nll, out);
}
